// RPN_87273735455067
// MI455X (gfx1250) — matrix operand order repaired and checked by reading, not compiled, not run on hardware
//
#include <hip/hip_runtime.h>
#include <math.h>

typedef __bf16 bf16;
typedef __attribute__((ext_vector_type(16))) __bf16 v16bf;
typedef __attribute__((ext_vector_type(8)))  float  v8f;

#define NB 8
#define NPIX 4096          // 64*64
#define NANCH 36864        // 4096*9
#define PRE_NMS 2000
#define POST_NMS 300
#define NWORDS 63          // ceil(2000/32)
#define LDSPAD 40          // 32 + 8 halves padding: 80B row, 16B aligned, conflict-free

union FragU { v16bf v; unsigned u[8]; };

// A fragment: lane holds row M=(lane&15); K halves per ISA 16-bit A 16x32 layout.
__device__ __forceinline__ v16bf frag_a(const bf16* rowp, int lane) {
  FragU f;
  int khi = (lane >> 4) << 3;                 // 0 or 8
#pragma unroll
  for (int g = 0; g < 2; ++g)
#pragma unroll
    for (int p = 0; p < 4; ++p)
      f.u[g * 4 + p] = *(const unsigned*)(rowp + g * 16 + khi + 2 * p);
  return f.v;
}
// B fragment: lane holds column N=(lane&15); K halves per ISA 16-bit layout (same as A).
__device__ __forceinline__ v16bf frag_b(const bf16* rowp, int lane) {
  FragU f;
  int khi = (lane >> 4) << 3;                 // 0 or 8
#pragma unroll
  for (int p = 0; p < 8; ++p)
    f.u[p] = *(const unsigned*)(rowp + (p >> 2) * 16 + khi + 2 * (p & 3));
  return f.v;
}

// ---------- weight prep: w1 fp32 [oc][ic][3][3] -> wb bf16 [f][oc][ic]; fused head weights ----------
__global__ void prep_weights(const float* __restrict__ w1,
                             const float* __restrict__ w_score, const float* __restrict__ b_score,
                             const float* __restrict__ w_loc,   const float* __restrict__ b_loc,
                             bf16* __restrict__ wb, bf16* __restrict__ wsl, float* __restrict__ bsl) {
  size_t tid = (size_t)blockIdx.x * 256 + threadIdx.x;
  if (tid < (size_t)9 * 512 * 512) {
    int ic = (int)(tid & 511);
    int oc = (int)((tid >> 9) & 511);
    int f  = (int)(tid >> 18);
    wb[tid] = (bf16)w1[((size_t)oc * 512 + ic) * 9 + f];
  }
  if (tid < 64 * 512) {
    int r = (int)(tid >> 9), k = (int)(tid & 511);
    float v = 0.f;
    if (r < 36) v = w_loc[r * 512 + k];
    else if (r < 54) v = w_score[(r - 36) * 512 + k];
    wsl[tid] = (bf16)v;
  }
  if (tid < 64) {
    int r = (int)tid;
    float v = 0.f;
    if (r < 36) v = b_loc[r];
    else if (r < 54) v = b_score[r - 36];
    bsl[tid] = v;
  }
}

// ---------- 3x3 conv as implicit GEMM: per batch [4096 x 4608] x [4608 x 512], bf16 WMMA ----------
__global__ __launch_bounds__(128) void conv3_wmma(const float* __restrict__ x,
                                                  const bf16* __restrict__ wb,
                                                  const float* __restrict__ b1,
                                                  bf16* __restrict__ h) {
  __shared__ bf16 As[64][LDSPAD];
  __shared__ bf16 Bs[64][LDSPAD];
  int bid = blockIdx.x;
  int nt = bid & 7, mt = (bid >> 3) & 63, n = bid >> 9;
  int oc0 = nt * 64;
  int y = mt;                       // tile = one image row y, x = 0..63
  int t = threadIdx.x;
  int lane = t & 31, wave = t >> 5;
  int wm = (wave >> 1) * 32, wn = (wave & 1) * 32;

  // loader mapping: thread owns row m = t>>1, 16 contiguous k at kb=(t&1)*16
  int lm = t >> 1;
  int kb = (t & 1) * 16;

  v8f acc[2][2] = {};

  for (int f = 0; f < 9; ++f) {
    int ky = f / 3, kx = f % 3;
    int yy = y + ky - 1;
    bool yok = (yy >= 0 && yy < 64);
    int yyc = yok ? yy : 0;
    int gx = lm + kx - 1;                       // per-thread uniform x with halo
    bool ok = yok && (gx >= 0) && (gx < 64);
    int cgx = gx < 0 ? 0 : (gx > 63 ? 63 : gx); // clamped, always safe
    const float* srcA = x + (((size_t)n * 512 + kb) * 64 + yyc) * 64 + cgx;
    const bf16* srcB = wb + (((size_t)f * 512 + oc0 + lm)) * 512 + kb;

    for (int c0 = 0; c0 < 512; c0 += 32) {
      { // A tile: branch-free loads (clamped addr + select), packed 16B LDS stores
        const float* sA = srcA + (size_t)c0 * 4096;
        __attribute__((aligned(16))) bf16 tmp[16];
#pragma unroll
        for (int i = 0; i < 16; ++i) {
          float v = sA[(size_t)i * 4096];       // unconditional, safe address
          tmp[i] = (bf16)(ok ? v : 0.f);
        }
        ((uint4*)&As[lm][kb])[0] = ((const uint4*)tmp)[0];
        ((uint4*)&As[lm][kb])[1] = ((const uint4*)tmp)[1];
      }
      { // B tile: 2x16B contiguous global loads -> 2x16B LDS stores
        const bf16* sB = srcB + c0;
        uint4 b0 = ((const uint4*)sB)[0];
        uint4 b1 = ((const uint4*)sB)[1];
        ((uint4*)&Bs[lm][kb])[0] = b0;
        ((uint4*)&Bs[lm][kb])[1] = b1;
      }
      if (c0 + 32 < 512) {                      // prefetch next K-chunk (global_prefetch_b8)
        __builtin_prefetch(srcA + (size_t)(c0 + 32) * 4096, 0, 1);
        __builtin_prefetch(srcB + c0 + 32, 0, 1);
      }
      __syncthreads();
      v16bf a[2], b[2];
#pragma unroll
      for (int i = 0; i < 2; ++i) a[i] = frag_a(&As[wm + i * 16 + (lane & 15)][0], lane);
#pragma unroll
      for (int j = 0; j < 2; ++j) b[j] = frag_b(&Bs[wn + j * 16 + (lane & 15)][0], lane);
#pragma unroll
      for (int i = 0; i < 2; ++i)
#pragma unroll
        for (int j = 0; j < 2; ++j)
          acc[i][j] = __builtin_amdgcn_wmma_f32_16x16x32_bf16(
              false, a[i], false, b[j], (short)0, acc[i][j], false, false);
      __syncthreads();
    }
  }
  int m0 = mt * 64;
#pragma unroll
  for (int i = 0; i < 2; ++i)
#pragma unroll
    for (int j = 0; j < 2; ++j) {
      int oc = oc0 + wn + j * 16 + (lane & 15);
      float bias = b1[oc];
#pragma unroll
      for (int v = 0; v < 8; ++v) {
        int m = m0 + wm + i * 16 + v + ((lane >> 4) << 3);
        float val = acc[i][j][v] + bias;
        h[((size_t)n * NPIX + m) * 512 + oc] = (bf16)fmaxf(val, 0.f);
      }
    }
}

// ---------- fused 1x1 convs: [4096 x 512] x [512 x 64] -> locs(36) + scores(18) ----------
__global__ __launch_bounds__(128) void head_wmma(const bf16* __restrict__ h,
                                                 const bf16* __restrict__ wsl,
                                                 const float* __restrict__ bsl,
                                                 float* __restrict__ out_locs,
                                                 float* __restrict__ out_scores) {
  __shared__ bf16 As[64][LDSPAD];
  __shared__ bf16 Bs[64][LDSPAD];
  int bid = blockIdx.x;
  int mt = bid & 63, n = bid >> 6;
  int m0 = mt * 64;
  int t = threadIdx.x, lane = t & 31, wave = t >> 5;
  int wm = (wave >> 1) * 32, wn = (wave & 1) * 32;
  int lm = t >> 1, kb = (t & 1) * 16;
  v8f acc[2][2] = {};

  const bf16* srcA = h + ((size_t)n * NPIX + m0 + lm) * 512 + kb;
  const bf16* srcB = wsl + (size_t)lm * 512 + kb;

  for (int c0 = 0; c0 < 512; c0 += 32) {
    {
      uint4 a0 = ((const uint4*)(srcA + c0))[0];
      uint4 a1 = ((const uint4*)(srcA + c0))[1];
      ((uint4*)&As[lm][kb])[0] = a0;
      ((uint4*)&As[lm][kb])[1] = a1;
    }
    {
      uint4 b0 = ((const uint4*)(srcB + c0))[0];
      uint4 b1 = ((const uint4*)(srcB + c0))[1];
      ((uint4*)&Bs[lm][kb])[0] = b0;
      ((uint4*)&Bs[lm][kb])[1] = b1;
    }
    if (c0 + 32 < 512) {
      __builtin_prefetch(srcA + c0 + 32, 0, 1);
      __builtin_prefetch(srcB + c0 + 32, 0, 1);
    }
    __syncthreads();
    v16bf a[2], b[2];
#pragma unroll
    for (int i = 0; i < 2; ++i) a[i] = frag_a(&As[wm + i * 16 + (lane & 15)][0], lane);
#pragma unroll
    for (int j = 0; j < 2; ++j) b[j] = frag_b(&Bs[wn + j * 16 + (lane & 15)][0], lane);
#pragma unroll
    for (int i = 0; i < 2; ++i)
#pragma unroll
      for (int j = 0; j < 2; ++j)
        acc[i][j] = __builtin_amdgcn_wmma_f32_16x16x32_bf16(
            false, a[i], false, b[j], (short)0, acc[i][j], false, false);
    __syncthreads();
  }
#pragma unroll
  for (int i = 0; i < 2; ++i)
#pragma unroll
    for (int j = 0; j < 2; ++j) {
      int col = wn + j * 16 + (lane & 15);
      float bias = bsl[col];
#pragma unroll
      for (int v = 0; v < 8; ++v) {
        int m = m0 + wm + i * 16 + v + ((lane >> 4) << 3);
        float val = acc[i][j][v] + bias;
        if (col < 36)       out_locs[((size_t)n * NPIX + m) * 36 + col] = val;
        else if (col < 54)  out_scores[((size_t)n * NPIX + m) * 18 + (col - 36)] = val;
      }
    }
}

// ---------- anchor + box decode + fg score + min-size mask ----------
__global__ void decode_kernel(const float* __restrict__ locs, const float* __restrict__ scores,
                              const int* __restrict__ img_h_p, const int* __restrict__ img_w_p,
                              float* __restrict__ boxes, float* __restrict__ msc,
                              float* __restrict__ anchor_out) {
  int tid = blockIdx.x * 256 + threadIdx.x;
  if (tid >= NB * NANCH) return;
  int n = tid / NANCH, i = tid - n * NANCH;
  int p = i / 9, a = i - p * 9;
  int xg = p >> 6, yg = p & 63;            // reference anchors are x-major: p = x*64 + y
  float sy = 16.f * yg, sx = 16.f * xg;
  const float ratios[3] = {0.5f, 1.f, 2.f};
  const float scales[3] = {8.f, 16.f, 32.f};
  int ri = a / 3, si = a - ri * 3;
  float hA = 16.f * scales[si] * sqrtf(ratios[ri]);
  float wA = 16.f * scales[si] * sqrtf(1.f / ratios[ri]);
  float ay1 = sy + 8.f - 0.5f * hA, ax1 = sx + 8.f - 0.5f * wA;
  float ay2 = sy + 8.f + 0.5f * hA, ax2 = sx + 8.f + 0.5f * wA;
  if (n == 0) {
    float* A = anchor_out + (size_t)i * 4;
    A[0] = ay1; A[1] = ax1; A[2] = ay2; A[3] = ax2;
  }
  float ah = ay2 - ay1, aw = ax2 - ax1;
  float acy = ay1 + 0.5f * ah, acx = ax1 + 0.5f * aw;
  const float* L = locs + ((size_t)n * NANCH + i) * 4;
  float dy = L[0], dx = L[1], dh = L[2], dw = L[3];
  float cy = dy * ah + acy, cx = dx * aw + acx;
  float bh = expf(dh) * ah, bw = expf(dw) * aw;
  float fh = (float)(*img_h_p), fw = (float)(*img_w_p);
  float y1 = fminf(fmaxf(cy - 0.5f * bh, 0.f), fh);
  float x1 = fminf(fmaxf(cx - 0.5f * bw, 0.f), fw);
  float y2 = fminf(fmaxf(cy + 0.5f * bh, 0.f), fh);
  float x2 = fminf(fmaxf(cx + 0.5f * bw, 0.f), fw);
  float* B = boxes + ((size_t)n * NANCH + i) * 4;
  B[0] = y1; B[1] = x1; B[2] = y2; B[3] = x2;
  const float* S = scores + ((size_t)n * NANCH + i) * 2;
  float fg = 1.f / (1.f + expf(-(S[1] - S[0])));   // softmax channel 1 == sigmoid(s1-s0)
  bool ok = (y2 - y1 >= 16.f) && (x2 - x1 >= 16.f);
  msc[(size_t)n * NANCH + i] = ok ? fg : -__builtin_inff();
}

__device__ __forceinline__ unsigned ordf(float f) {
  unsigned u = __float_as_uint(f);
  return (u & 0x80000000u) ? ~u : (u | 0x80000000u);
}

// ---------- exact top-2000 per batch: rank-select via 64-bit key bisection + bitonic sort ----------
__global__ __launch_bounds__(1024) void topk_kernel(const float* __restrict__ msc,
                                                    const float* __restrict__ boxes,
                                                    int* __restrict__ sel, float* __restrict__ sb,
                                                    int* __restrict__ valid) {
  __shared__ unsigned long long sk[2048];
  __shared__ int cnt;
  int n = blockIdx.x, t = threadIdx.x;
  const float* sc = msc + (size_t)n * NANCH;
  unsigned long long key[36];
#pragma unroll
  for (int s = 0; s < 36; ++s) {
    int i = t + s * 1024;
    key[s] = ((unsigned long long)ordf(sc[i]) << 32) | (unsigned)(0xFFFFFFFFu - (unsigned)i);
  }
  unsigned long long lo = 0, hi = ~0ull;
  while (lo < hi) {                                  // <=64 iterations
    unsigned long long mid = lo + ((hi - lo) >> 1) + 1;
    if (t == 0) cnt = 0;
    __syncthreads();
    int c = 0;
#pragma unroll
    for (int s = 0; s < 36; ++s) c += (key[s] >= mid) ? 1 : 0;
    if (c) atomicAdd(&cnt, c);
    __syncthreads();
    int total = cnt;
    __syncthreads();
    if (total >= PRE_NMS) lo = mid; else hi = mid - 1;
  }
  unsigned long long T = lo;                          // keys unique -> exactly 2000 >= T
  if (t == 0) cnt = 0;
  sk[t] = 0; sk[t + 1024] = 0;
  __syncthreads();
#pragma unroll
  for (int s = 0; s < 36; ++s)
    if (key[s] >= T) { int pos = atomicAdd(&cnt, 1); if (pos < 2048) sk[pos] = key[s]; }
  __syncthreads();
  for (int k = 2; k <= 2048; k <<= 1) {              // bitonic sort, descending
    for (int j = k >> 1; j > 0; j >>= 1) {
#pragma unroll
      for (int r = 0; r < 2; ++r) {
        int idx = t + r * 1024;
        int ixj = idx ^ j;
        if (ixj > idx) {
          bool desc = (idx & k) == 0;
          unsigned long long A = sk[idx], B = sk[ixj];
          if (desc ? (A < B) : (A > B)) { sk[idx] = B; sk[ixj] = A; }
        }
      }
      __syncthreads();
    }
  }
  for (int r = t; r < PRE_NMS; r += 1024) {
    unsigned long long kk = sk[r];
    unsigned idx = 0xFFFFFFFFu - (unsigned)(kk & 0xFFFFFFFFu);
    sel[(size_t)n * PRE_NMS + r] = (int)idx;
    const float* B = boxes + ((size_t)n * NANCH + idx) * 4;
    float* O = sb + ((size_t)n * PRE_NMS + r) * 4;
    O[0] = B[0]; O[1] = B[1]; O[2] = B[2]; O[3] = B[3];
    valid[(size_t)n * PRE_NMS + r] = (sc[idx] > -1e30f) ? 1 : 0;
  }
}

// ---------- NMS suppression bit-matrix: row i, 63 words over j ----------
__global__ __launch_bounds__(64) void nms_mask_kernel(const float* __restrict__ sb,
                                                      unsigned* __restrict__ mask) {
  int bid = blockIdx.x;
  int n = bid / PRE_NMS, i = bid - n * PRE_NMS;
  int t = threadIdx.x;
  if (t >= NWORDS) return;
  const float* bi = sb + ((size_t)n * PRE_NMS + i) * 4;
  float y1 = bi[0], x1 = bi[1], y2 = bi[2], x2 = bi[3];
  float area_i = (y2 - y1) * (x2 - x1);
  unsigned bits = 0;
  for (int b = 0; b < 32; ++b) {
    int j = t * 32 + b;
    if (j < PRE_NMS) {
      const float* bj = sb + ((size_t)n * PRE_NMS + j) * 4;
      float jy1 = bj[0], jx1 = bj[1], jy2 = bj[2], jx2 = bj[3];
      float area_j = (jy2 - jy1) * (jx2 - jx1);
      float ty = fmaxf(y1, jy1), tx = fmaxf(x1, jx1);
      float by = fminf(y2, jy2), bx = fminf(x2, jx2);
      float inter = fmaxf(by - ty, 0.f) * fmaxf(bx - tx, 0.f);
      float iou = inter / (area_i + area_j - inter + 1e-10f);
      if (iou > 0.7f) bits |= (1u << b);
    }
  }
  mask[((size_t)n * PRE_NMS + i) * NWORDS + t] = bits;
}

// ---------- sequential greedy scan (matches dynamic-keep reference), emit rois ----------
__global__ __launch_bounds__(64) void nms_scan_kernel(const unsigned* __restrict__ mask,
                                                      const float* __restrict__ sb,
                                                      const int* __restrict__ valid,
                                                      float* __restrict__ rois,
                                                      float* __restrict__ roi_idx) {
  __shared__ unsigned rem[NWORDS];
  __shared__ int keepcnt;
  int n = blockIdx.x, t = threadIdx.x;
  if (t < NWORDS) {
    unsigned r = 0;
    for (int b = 0; b < 32; ++b) {
      int j = t * 32 + b;
      if (j >= PRE_NMS || !valid[(size_t)n * PRE_NMS + j]) r |= (1u << b);
    }
    rem[t] = r;
  }
  if (t == 0) keepcnt = 0;
  for (int r = t; r < POST_NMS; r += 64) {
    float* R = rois + ((size_t)n * POST_NMS + r) * 4;
    R[0] = R[1] = R[2] = R[3] = 0.f;
    roi_idx[(size_t)n * POST_NMS + r] = (float)n;
  }
  __syncthreads();
  for (int i = 0; i < PRE_NMS; ++i) {
    bool alive = ((rem[i >> 5] >> (i & 31)) & 1u) == 0u;
    __syncthreads();
    if (alive) {
      if (t < NWORDS) {
        int iw = i >> 5, ib = i & 31;
        unsigned mw = mask[((size_t)n * PRE_NMS + i) * NWORDS + t];
        if (t < iw) mw = 0;
        else if (t == iw) {
          unsigned lowmask = (ib == 31) ? 0xFFFFFFFFu : ((2u << ib) - 1u);
          mw &= ~lowmask;                 // only suppress j > i
        }
        rem[t] |= mw;
      }
      if (t == 0) {
        int r = keepcnt;
        if (r < POST_NMS) {
          const float* B = sb + ((size_t)n * PRE_NMS + i) * 4;
          float* R = rois + ((size_t)n * POST_NMS + r) * 4;
          R[0] = B[0]; R[1] = B[1]; R[2] = B[2]; R[3] = B[3];
        }
        keepcnt = r + 1;
      }
    }
    __syncthreads();
  }
}

extern "C" void kernel_launch(void* const* d_in, const int* in_sizes, int n_in,
                              void* d_out, int out_size, void* d_ws, size_t ws_size,
                              hipStream_t stream) {
  const float* x       = (const float*)d_in[0];
  const float* w1      = (const float*)d_in[1];
  const float* b1      = (const float*)d_in[2];
  const float* w_score = (const float*)d_in[3];
  const float* b_score = (const float*)d_in[4];
  const float* w_loc   = (const float*)d_in[5];
  const float* b_loc   = (const float*)d_in[6];
  const int*   img_h   = (const int*)d_in[7];
  const int*   img_w   = (const int*)d_in[8];
  (void)in_sizes; (void)n_in; (void)out_size; (void)ws_size;

  float* out        = (float*)d_out;                 // tuple concatenated flat
  float* out_locs   = out;                                           // 8*36864*4
  float* out_scores = out_locs + (size_t)NB * NANCH * 4;             // 8*36864*2
  float* out_rois   = out_scores + (size_t)NB * NANCH * 2;           // 8*300*4
  float* out_ridx   = out_rois + (size_t)NB * POST_NMS * 4;          // 8*300
  float* out_anchor = out_ridx + (size_t)NB * POST_NMS;              // 36864*4

  char* ws = (char*)d_ws;
  size_t off = 0;
  auto alloc = [&](size_t bytes) { void* p = ws + off; off += (bytes + 255) & ~(size_t)255; return p; };
  bf16*  wb    = (bf16*)alloc((size_t)9 * 512 * 512 * 2);
  bf16*  wsl   = (bf16*)alloc((size_t)64 * 512 * 2);
  float* bsl   = (float*)alloc(64 * 4);
  bf16*  h     = (bf16*)alloc((size_t)NB * NPIX * 512 * 2);
  float* boxes = (float*)alloc((size_t)NB * NANCH * 4 * 4);
  float* msc   = (float*)alloc((size_t)NB * NANCH * 4);
  int*   sel   = (int*)alloc((size_t)NB * PRE_NMS * 4);
  float* sb    = (float*)alloc((size_t)NB * PRE_NMS * 4 * 4);
  int*   valid = (int*)alloc((size_t)NB * PRE_NMS * 4);
  unsigned* mk = (unsigned*)alloc((size_t)NB * PRE_NMS * NWORDS * 4);

  prep_weights<<<(9 * 512 * 512 + 255) / 256, 256, 0, stream>>>(w1, w_score, b_score, w_loc, b_loc, wb, wsl, bsl);
  conv3_wmma<<<NB * 64 * 8, 128, 0, stream>>>(x, wb, b1, h);
  head_wmma<<<NB * 64, 128, 0, stream>>>(h, wsl, bsl, out_locs, out_scores);
  decode_kernel<<<(NB * NANCH + 255) / 256, 256, 0, stream>>>(out_locs, out_scores, img_h, img_w, boxes, msc, out_anchor);
  topk_kernel<<<NB, 1024, 0, stream>>>(msc, boxes, sel, sb, valid);
  nms_mask_kernel<<<NB * PRE_NMS, 64, 0, stream>>>(sb, mk);
  nms_scan_kernel<<<NB, 64, 0, stream>>>(mk, sb, valid, out_rois, out_ridx);
}
